// MultiHeadAttention_12764642803908
// MI455X (gfx1250) — matrix operand order repaired and checked by reading, not compiled, not run on hardware
//
#include <hip/hip_runtime.h>

// ---------------------------------------------------------------------------
// MI455X (gfx1250) multi-head attention, bf16 WMMA path, double-buffered LDS,
// async global->LDS staging (GLOBAL_LOAD_ASYNC_TO_LDS_B128, ASYNCcnt-tracked)
// where no format conversion is needed.
// B=8, S=1024, E=768, H=12, D=64.
// ---------------------------------------------------------------------------

typedef __attribute__((ext_vector_type(16))) __bf16 v16bf;
typedef __attribute__((ext_vector_type(8)))  __bf16 v8bf;
typedef __attribute__((ext_vector_type(4)))  __bf16 v4bf;
typedef __attribute__((ext_vector_type(8)))  float  v8f;
typedef __attribute__((ext_vector_type(4)))  float  v4f;
typedef __attribute__((ext_vector_type(4)))  int    v4i;
typedef __attribute__((ext_vector_type(4)))  unsigned int v4u;

#define WMMA_BF16(a, b, c) \
  __builtin_amdgcn_wmma_f32_16x16x32_bf16(false, (a), false, (b), (short)0, (c), false, false)

static constexpr int kB = 8, kS = 1024, kE = 768, kH = 12, kD = 64;
static constexpr int kRows = kB * kS;   // 8192

#if __has_builtin(__builtin_amdgcn_global_load_async_to_lds_b128)
#define ATHENA_ASYNC_LDS 1
typedef __attribute__((address_space(1))) v4i* as1_v4i;
typedef __attribute__((address_space(3))) v4i* as3_v4i;
#endif

// 16-byte global -> LDS copy; async (ASYNCcnt) when the gfx1250 builtin exists.
__device__ __forceinline__ void async_copy16(const void* g, void* l) {
#ifdef ATHENA_ASYNC_LDS
  v4i* gp = (v4i*)g;                       // strip const in generic AS
  v4i* lp = (v4i*)l;
  __builtin_amdgcn_global_load_async_to_lds_b128((as1_v4i)gp, (as3_v4i)lp, 0, 0);
#else
  *(v4u*)l = *(const v4u*)g;
#endif
}

__device__ __forceinline__ void wait_async0() {
#ifdef ATHENA_ASYNC_LDS
#if __has_builtin(__builtin_amdgcn_s_wait_asynccnt)
  __builtin_amdgcn_s_wait_asynccnt(0);
#else
  asm volatile("s_wait_asynccnt 0" ::: "memory");
#endif
#endif
}

__device__ __forceinline__ v16bf frag_pair(const __bf16* p0, const __bf16* p1) {
  v8bf lo = *(const v8bf*)p0;
  v8bf hi = *(const v8bf*)p1;
  v16bf r;
#pragma unroll
  for (int i = 0; i < 8; ++i) { r[i] = lo[i]; r[8 + i] = hi[i]; }
  return r;
}

__device__ __forceinline__ v4bf to_bf4(v4f x) {
  v4bf y;
#pragma unroll
  for (int e = 0; e < 4; ++e) y[e] = (__bf16)x[e];
  return y;
}

// ---------------------------------------------------------------------------
// GEMM: Y = A (8192 x 768) @ W^T, W (768 x 768) row-major [n][k] f32.
// Block: 256 threads (8 waves). Tile M=256 (wave strips of 32), N=64, Kstep=32.
// Double-buffered LDS, one barrier per K-iteration.
// HEAD_OUT: write oscale*Y as bf16 in (b, h=blockIdx.y, s, d); else f32.
// ---------------------------------------------------------------------------
template <bool A_BF16, bool HEAD_OUT>
__global__ __launch_bounds__(256) void gemm_xwT(const void* __restrict__ Aptr,
                                                const float* __restrict__ W,
                                                void* __restrict__ Yptr,
                                                float oscale) {
  constexpr int LDA = 40;            // padded bf16 stride: conflict-free 16B reads
  constexpr int NK  = kE / 32;       // 24
  __shared__ __bf16 lA[2][256 * LDA];
  __shared__ __bf16 lB[2][64 * LDA];

  const int t    = threadIdx.x;
  const int lane = t & 31, wv = t >> 5;
  const int half = lane >> 4, ln = lane & 15;
  const int m0   = blockIdx.x * 256;
  const int nb   = blockIdx.y;

  const float*  Af = (const float*)Aptr;
  const __bf16* Ab = (const __bf16*)Aptr;

  v8f acc[2][4] = {};

  // ---- prologue: stage tile 0 into buffer 0 ----
  if (A_BF16) {
#pragma unroll
    for (int j = 0; j < 4; ++j) {            // 1024 granules of 16B
      int id = j * 256 + t, row = id >> 2, c16 = (id & 3) * 8;
      async_copy16(Ab + (size_t)(m0 + row) * kE + c16, &lA[0][row * LDA + c16]);
    }
  } else {
#pragma unroll
    for (int j = 0; j < 8; ++j) {            // 2048 groups of 4 f32
      int id = j * 256 + t, row = id >> 3, kq = (id & 7) * 4;
      v4f x = *(const v4f*)(Af + (size_t)(m0 + row) * kE + kq);
      *(v4bf*)&lA[0][row * LDA + kq] = to_bf4(x);
    }
  }
#pragma unroll
  for (int j = 0; j < 2; ++j) {              // W tile 64x32
    int id = j * 256 + t, row = id >> 3, kq = (id & 7) * 4;
    v4f x = *(const v4f*)(W + (size_t)(nb * 64 + row) * kE + kq);
    *(v4bf*)&lB[0][row * LDA + kq] = to_bf4(x);
  }
  wait_async0();
  __syncthreads();

  for (int kt = 0; kt < NK; ++kt) {
    const int  cur = kt & 1, nxt = cur ^ 1;
    const int  kn  = (kt + 1) * 32;
    const bool have_next = (kt + 1) < NK;

    // ---- issue next tile's global traffic before compute ----
    v4f aA[8], aW[2];
    if (have_next) {
      if (A_BF16) {
#pragma unroll
        for (int j = 0; j < 4; ++j) {
          int id = j * 256 + t, row = id >> 2, c16 = (id & 3) * 8;
          async_copy16(Ab + (size_t)(m0 + row) * kE + kn + c16,
                       &lA[nxt][row * LDA + c16]);
        }
      } else {
#pragma unroll
        for (int j = 0; j < 8; ++j) {
          int id = j * 256 + t, row = id >> 3, kq = (id & 7) * 4;
          aA[j] = *(const v4f*)(Af + (size_t)(m0 + row) * kE + kn + kq);
        }
      }
#pragma unroll
      for (int j = 0; j < 2; ++j) {
        int id = j * 256 + t, row = id >> 3, kq = (id & 7) * 4;
        aW[j] = *(const v4f*)(W + (size_t)(nb * 64 + row) * kE + kn + kq);
      }
    }

    // ---- compute: wave strip rows [wv*32, wv*32+32), 4 N chunks, 8 WMMA ----
    const __bf16* a0 = &lA[cur][(wv * 32 + ln) * LDA];
    const __bf16* a1 = &lA[cur][(wv * 32 + 16 + ln) * LDA];
    v16bf af0 = frag_pair(a0 + half * 8, a0 + 16 + half * 8);
    v16bf af1 = frag_pair(a1 + half * 8, a1 + 16 + half * 8);
#pragma unroll
    for (int c = 0; c < 4; ++c) {
      const __bf16* brow = &lB[cur][(c * 16 + ln) * LDA + half * 8];
      v16bf bf = frag_pair(brow, brow + 16);
      acc[0][c] = WMMA_BF16(af0, bf, acc[0][c]);
      acc[1][c] = WMMA_BF16(af1, bf, acc[1][c]);
    }

    // ---- write staged registers into the other buffer ----
    if (have_next) {
      if (!A_BF16) {
#pragma unroll
        for (int j = 0; j < 8; ++j) {
          int id = j * 256 + t, row = id >> 3, kq = (id & 7) * 4;
          *(v4bf*)&lA[nxt][row * LDA + kq] = to_bf4(aA[j]);
        }
      }
#pragma unroll
      for (int j = 0; j < 2; ++j) {
        int id = j * 256 + t, row = id >> 3, kq = (id & 7) * 4;
        *(v4bf*)&lB[nxt][row * LDA + kq] = to_bf4(aW[j]);
      }
    }
    wait_async0();
    __syncthreads();
  }

  // ---- epilogue: C layout lane holds col n = ln, rows r + 8*half ----
#pragma unroll
  for (int sp = 0; sp < 2; ++sp) {
    if (HEAD_OUT) {
      __bf16* Y = (__bf16*)Yptr;
#pragma unroll
      for (int c = 0; c < 4; ++c) {
        int d = c * 16 + ln;
#pragma unroll
        for (int r = 0; r < 8; ++r) {
          int row = m0 + wv * 32 + sp * 16 + r + 8 * half;
          int b = row >> 10, s = row & 1023;
          Y[(((size_t)b * kH + nb) * kS + s) * kD + d] =
              (__bf16)(acc[sp][c][r] * oscale);
        }
      }
    } else {
      float* Y = (float*)Yptr;
#pragma unroll
      for (int c = 0; c < 4; ++c) {
        int n = nb * 64 + c * 16 + ln;
#pragma unroll
        for (int r = 0; r < 8; ++r) {
          int row = m0 + wv * 32 + sp * 16 + r + 8 * half;
          Y[(size_t)row * kE + n] = acc[sp][c][r];
        }
      }
    }
  }
}

// ---------------------------------------------------------------------------
// Causal flash attention, transposed-scores trick, double-buffered K/V tiles.
// Block: 128 threads (4 waves), one (b,h), 64 q rows (16 per wave).
// Q arrives pre-scaled by 1/sqrt(D).  K staged [kv][d] (async copy),
// V staged transposed [d][kv] (register path).  Causal masking only on
// diagonal tiles (wave-uniform branch).
// ---------------------------------------------------------------------------
__global__ __launch_bounds__(128) void attn_kernel(const __bf16* __restrict__ q,
                                                   const __bf16* __restrict__ k,
                                                   const __bf16* __restrict__ v,
                                                   __bf16* __restrict__ ctx) {
  constexpr int LDK = 72;   // 144 B rows
  constexpr int LDV = 40;   // 80 B rows
  __shared__ __bf16 lK[2][32 * LDK];
  __shared__ __bf16 lV[2][64 * LDV];

  const int t = threadIdx.x, lane = t & 31, wv = t >> 5;
  const int half = lane >> 4, ln = lane & 15;
  const int bh = blockIdx.y, b = bh / kH, h = bh % kH;
  const int q0 = blockIdx.x * 64;
  const int qw = q0 + wv * 16;
  const size_t base = (size_t)bh * kS * kD;

  // Q as B-fragments (fixed): lane col = q row qw+ln, frag c: d = c*32+half*8..+8 and c*32+16+half*8..+8
  v16bf bq[2];
#pragma unroll
  for (int c = 0; c < 2; ++c) {
    const __bf16* p = q + base + (size_t)(qw + ln) * kD + c * 32 + half * 8;
    bq[c] = frag_pair(p, p + 16);
  }

  v8f o[4] = {};
  float mrow = -1e30f, lrow = 0.f;
  const int ntiles = q0 / 32 + 2;              // causal: kv <= q0+63

  auto stageK = [&](int kvb, int buf) {
#pragma unroll
    for (int j = 0; j < 2; ++j) {              // 256 granules of 16B
      int id = j * 128 + t, row = id >> 3, dq = (id & 7) * 8;
      async_copy16(k + base + (size_t)(kvb + row) * kD + dq,
                   &lK[buf][row * LDK + dq]);
    }
  };
  auto loadV = [&](int kvb, v4bf* vr) {
#pragma unroll
    for (int j = 0; j < 4; ++j) {              // 512 groups of 4 bf16
      int id = j * 128 + t, row = id >> 4, dq = (id & 15) * 4;
      vr[j] = *(const v4bf*)(v + base + (size_t)(kvb + row) * kD + dq);
    }
  };
  auto scatterV = [&](const v4bf* vr, int buf) {
#pragma unroll
    for (int j = 0; j < 4; ++j) {
      int id = j * 128 + t, row = id >> 4, dq = (id & 15) * 4;
#pragma unroll
      for (int e = 0; e < 4; ++e) lV[buf][(dq + e) * LDV + row] = vr[j][e];
    }
  };

  {                                            // prologue: tile 0 -> buffer 0
    v4bf vr[4];
    stageK(0, 0);
    loadV(0, vr);
    scatterV(vr, 0);
  }
  wait_async0();
  __syncthreads();

  for (int tt = 0; tt < ntiles; ++tt) {
    const int  cur = tt & 1, nxt = cur ^ 1;
    const int  kvb = tt * 32;
    const bool have_next = (tt + 1) < ntiles;

    v4bf vr[4];
    if (have_next) { stageK((tt + 1) * 32, nxt); loadV((tt + 1) * 32, vr); }

    if (kvb <= qw + 15) {                      // wave-uniform causal skip
      // S^T = K @ Q^T: T0 = kv rows 0..15, T1 = kv rows 16..31.
      v8f t0 = {}, t1 = {};
#pragma unroll
      for (int c = 0; c < 2; ++c) {
        const __bf16* a0 = &lK[cur][ln * LDK + c * 32 + half * 8];
        t0 = WMMA_BF16(frag_pair(a0, a0 + 16), bq[c], t0);
        const __bf16* a1 = &lK[cur][(16 + ln) * LDK + c * 32 + half * 8];
        t1 = WMMA_BF16(frag_pair(a1, a1 + 16), bq[c], t1);
      }

      // Masking needed only on diagonal tiles (wave-uniform predicate).
      const bool diag = (kvb + 31 > qw);
      const int  qcol = qw + ln;
      float tmax = -1e30f;
      if (diag) {
#pragma unroll
        for (int r = 0; r < 8; ++r) {
          int kv0 = kvb + r + 8 * half, kv1 = kv0 + 16;
          float s0 = (kv0 > qcol) ? -1e30f : t0[r];
          float s1 = (kv1 > qcol) ? -1e30f : t1[r];
          t0[r] = s0; t1[r] = s1;
          tmax = fmaxf(tmax, fmaxf(s0, s1));
        }
      } else {
#pragma unroll
        for (int r = 0; r < 8; ++r)
          tmax = fmaxf(tmax, fmaxf(t0[r], t1[r]));
      }
      tmax = fmaxf(tmax, __shfl_xor(tmax, 16, 32));
      float mnew  = fmaxf(mrow, tmax);
      float alpha = __expf(mrow - mnew);
      mrow = mnew;

      float psum = 0.f;
      if (diag) {
#pragma unroll
        for (int r = 0; r < 8; ++r) {
          int kv0 = kvb + r + 8 * half, kv1 = kv0 + 16;
          float p0 = (kv0 > qcol) ? 0.f : __expf(t0[r] - mnew);
          float p1 = (kv1 > qcol) ? 0.f : __expf(t1[r] - mnew);
          t0[r] = p0; t1[r] = p1; psum += p0 + p1;
        }
      } else {
#pragma unroll
        for (int r = 0; r < 8; ++r) {
          float p0 = __expf(t0[r] - mnew);
          float p1 = __expf(t1[r] - mnew);
          t0[r] = p0; t1[r] = p1; psum += p0 + p1;
        }
      }
      psum += __shfl_xor(psum, 16, 32);
      lrow = lrow * alpha + psum;

      // P^T in C-layout == P in A-fragment layout: direct repack.
      v16bf ap;
#pragma unroll
      for (int e = 0; e < 8; ++e) { ap[e] = (__bf16)t0[e]; ap[8 + e] = (__bf16)t1[e]; }

#pragma unroll
      for (int r = 0; r < 8; ++r) {
        float ab = __shfl(alpha, r + 8 * half, 32);
#pragma unroll
        for (int c = 0; c < 4; ++c) o[c][r] *= ab;
      }
#pragma unroll
      for (int c = 0; c < 4; ++c) {
        const __bf16* bp = &lV[cur][(c * 16 + ln) * LDV + half * 8];
        o[c] = WMMA_BF16(ap, frag_pair(bp, bp + 16), o[c]);
      }
    }

    if (have_next) scatterV(vr, nxt);
    wait_async0();
    __syncthreads();
  }

  // finalize: divide O rows by l, write ctx in (b, s, h*64+d) bf16 layout.
  float linv = 1.f / lrow;
#pragma unroll
  for (int r = 0; r < 8; ++r) {
    float li = __shfl(linv, r + 8 * half, 32);
    int qrow = qw + r + 8 * half;
#pragma unroll
    for (int c = 0; c < 4; ++c) {
      int d = c * 16 + ln;
      ctx[((size_t)(b * kS + qrow)) * kE + h * kD + d] = (__bf16)(o[c][r] * li);
    }
  }
}

// ---------------------------------------------------------------------------
extern "C" void kernel_launch(void* const* d_in, const int* in_sizes, int n_in,
                              void* d_out, int out_size, void* d_ws, size_t ws_size,
                              hipStream_t stream) {
  const float* x  = (const float*)d_in[0];
  const float* wq = (const float*)d_in[1];
  const float* wk = (const float*)d_in[2];
  const float* wv = (const float*)d_in[3];
  const float* wo = (const float*)d_in[4];

  const size_t qsz = (size_t)kRows * kE;       // elements per bf16 buffer
  __bf16* qws = (__bf16*)d_ws;                 // needs 4 * 12.6 MB ~= 50 MB ws
  __bf16* kws = qws + qsz;
  __bf16* vws = kws + qsz;
  __bf16* cws = vws + qsz;

  dim3 gblk(256), ggrd(kRows / 256, kE / 64, 1);
  // Q pre-scaled by 1/sqrt(D) so attention needs no per-score multiply.
  gemm_xwT<false, true><<<ggrd, gblk, 0, stream>>>((const void*)x, wq, (void*)qws, 0.125f);
  gemm_xwT<false, true><<<ggrd, gblk, 0, stream>>>((const void*)x, wk, (void*)kws, 1.0f);
  gemm_xwT<false, true><<<ggrd, gblk, 0, stream>>>((const void*)x, wv, (void*)vws, 1.0f);

  attn_kernel<<<dim3(kS / 64, kB * kH), 128, 0, stream>>>(qws, kws, vws, cws);

  gemm_xwT<true, false><<<ggrd, gblk, 0, stream>>>((const void*)cws, wo, d_out, 1.0f);
}
